// Attention_70136815943694
// MI455X (gfx1250) — matrix operand order repaired and checked by reading, not compiled, not run on hardware
//
#include <hip/hip_runtime.h>

// Problem constants (match reference)
#define DMODEL 1024
#define SLEN   2048
#define BATCH  2
#define NHEADS 16
#define HDIM   64

// KV staging mechanism: 1 = TDM tensor_load_to_lds (one DMA per 32-key block),
// 0 = per-lane global_load_async_to_lds_b128 (round-3 fallback).
#define USE_TDM 1

typedef __attribute__((ext_vector_type(16))) __bf16        bfx16;
typedef __attribute__((ext_vector_type(8)))  float         fx8;
typedef __attribute__((ext_vector_type(4)))  float         fx4;
typedef __attribute__((ext_vector_type(4)))  unsigned int  ux4;
typedef __attribute__((ext_vector_type(8)))  unsigned int  ux8;

union Frag {
    unsigned int  u[8];
    ux4           u4[2];
    bfx16         bf;
};

__device__ __forceinline__ unsigned short f2bf(float f) {
    unsigned int u = __float_as_uint(f);
    u += 0x7FFFu + ((u >> 16) & 1u);      // round-to-nearest-even truncation
    return (unsigned short)(u >> 16);
}
__device__ __forceinline__ unsigned int pk2(float a, float b) {
    return (unsigned int)f2bf(a) | ((unsigned int)f2bf(b) << 16);
}

// Async global->LDS copy of 16B per lane (ASYNCcnt path, ISA 15.18.3).
__device__ __forceinline__ void async_copy_b128(unsigned int lds_off,
                                                const void* gaddr) {
    asm volatile("global_load_async_to_lds_b128 %0, %1, off"
                 :: "v"(lds_off), "v"((unsigned long long)(uintptr_t)gaddr)
                 : "memory");
}

#if USE_TDM
// TDM: one wave-level DMA of a 2D tile (32 rows x 128B) global -> LDS.
// D# groups per ISA ch8: group0 = {count/lds_addr/global_addr/type},
// group1 = {mask, data_size, dims/strides}. 2-group form (<=2D tensor).
__device__ __forceinline__ void tdm_stage_block(unsigned int lds_byte_off,
                                                const void* gaddr) {
    const unsigned long long ga = (unsigned long long)(uintptr_t)gaddr;
    ux4 g0;
    g0[0] = (unsigned int)__builtin_amdgcn_readfirstlane(1u);  // count=1 (valid)
    g0[1] = (unsigned int)__builtin_amdgcn_readfirstlane(lds_byte_off);
    g0[2] = (unsigned int)__builtin_amdgcn_readfirstlane((unsigned int)ga);
    g0[3] = (unsigned int)__builtin_amdgcn_readfirstlane(
                ((unsigned int)(ga >> 32) & 0x01FFFFFFu) | (2u << 30)); // type=2
    ux8 g1;
    g1[0] = 3u << 16;     // workgroup_mask=0 (not in cluster), data_size=3 (8B)
    g1[1] = 16u << 16;    // tensor_dim0 = 16   (16 x 8B = 128B row)
    g1[2] = 32u << 16;    // tensor_dim0 hi=0, tensor_dim1 = 32 rows
    g1[3] = 16u << 16;    // tensor_dim1 hi=0, tile_dim0 = 16
    g1[4] = 32u;          // tile_dim1 = 32, tile_dim2 = 0
    g1[5] = 16u;          // tensor_dim0_stride = 16 (8B units)
    g1[6] = 0u;           // stride hi / tensor_dim1_stride lo
    g1[7] = 0u;           // tensor_dim1_stride hi
    asm volatile("tensor_load_to_lds %0, %1" :: "s"(g0), "s"(g1) : "memory");
}
#endif

// ---------------------------------------------------------------------------
// One-pass f32 -> bf16 conversion (8 elements / thread, packed 16B stores).
// ---------------------------------------------------------------------------
__global__ void __launch_bounds__(256)
cvt_bf16_kernel(const float* __restrict__ in, unsigned short* __restrict__ out,
                int n8)
{
    const int i = blockIdx.x * 256 + threadIdx.x;
    if (i >= n8) return;
    const fx4* p = (const fx4*)(in + (size_t)i * 8);
    fx4 a = p[0], b = p[1];
    ux4 o;
    o[0] = pk2(a[0], a[1]); o[1] = pk2(a[2], a[3]);
    o[2] = pk2(b[0], b[1]); o[3] = pk2(b[2], b[3]);
    *(ux4*)(out + (size_t)i * 8) = o;
}

// ---------------------------------------------------------------------------
// Projection: Out_headmajor[b*16+h][s][i] = bf16( x[r,:] . W[c,:] + bias[c] )
// where r = b*SLEN+s, c = i*16 + h  (head index fast-varying, per reference).
// One wave = 16x64 tile, double-buffered fragments; latency bubbles are
// covered by ~16 waves/SIMD of occupancy (4096 waves total).
// ---------------------------------------------------------------------------
__global__ void __launch_bounds__(256)
proj_kernel(const unsigned short* __restrict__ Xb,
            const unsigned short* __restrict__ Wb,
            const float* __restrict__ bias, unsigned short* __restrict__ Out)
{
    const int lane  = threadIdx.x & 31;
    const int wglob = blockIdx.x * 8 + (threadIdx.x >> 5);
    const int rt    = wglob >> 4;      // row tile 0..255
    const int cg    = wglob & 15;      // 64-col group 0..15
    const int half  = lane >> 4;
    const int l16   = lane & 15;

    const int aoff = half ? 16 : 0;    // A lane k-pattern: {0..15}/{16..31}
    const int boff = half ? 16 : 0;    // B lane k-pattern: {0..15}/{16..31}

    const unsigned short* xrow  = Xb + (size_t)(rt * 16 + l16) * DMODEL;
    const unsigned short* wbase = Wb + (size_t)(cg*64 + l16) * DMODEL + boff;

    const fx8 zero8 = {0.f,0.f,0.f,0.f,0.f,0.f,0.f,0.f};
    fx8 acc[4];
    #pragma unroll
    for (int t = 0; t < 4; ++t) acc[t] = zero8;

    Frag A[2], Bf[2][4];
    auto load_step = [&](int k, int s) {
        A[s].u4[0] = *(const ux4*)(xrow + k + aoff);
        A[s].u4[1] = *(const ux4*)(xrow + k + aoff + 8);
        #pragma unroll
        for (int nt = 0; nt < 4; ++nt) {
            const unsigned short* wr = wbase + (size_t)nt * 16 * DMODEL + k;
            Bf[s][nt].u4[0] = *(const ux4*)(wr);
            Bf[s][nt].u4[1] = *(const ux4*)(wr + 8);
        }
    };

    load_step(0, 0);
    int s = 0;
    for (int k = 0; k < DMODEL; k += 32, s ^= 1) {
        if (k + 32 < DMODEL) load_step(k + 32, s ^ 1);
        #pragma unroll
        for (int nt = 0; nt < 4; ++nt)
            acc[nt] = __builtin_amdgcn_wmma_f32_16x16x32_bf16(
                false, A[s].bf, false, Bf[s][nt].bf, (short)0, acc[nt],
                false, false);
    }

    // Epilogue: bias add + scatter to head-major bf16 [B*NH][SLEN][HDIM]
    #pragma unroll
    for (int nt = 0; nt < 4; ++nt) {
        const int c = cg*64 + nt*16 + l16;
        const float bval = bias[c];
        const int h = c & 15;    // head (fast-varying channel index)
        const int i = c >> 4;    // dim within head
        #pragma unroll
        for (int r = 0; r < 8; ++r) {
            const int row  = rt*16 + r + half*8;
            const int bidx = row >> 11;       // / SLEN
            const int sPos = row & (SLEN-1);
            const size_t o = (((size_t)(bidx*NHEADS + h) * SLEN) + sPos) * HDIM + i;
            Out[o] = f2bf(acc[nt][r] + bval);
        }
    }
}

// ---------------------------------------------------------------------------
// Flash attention (K == V per the reference's faithful bug). One wave owns a
// 16-query tile of one (batch, head). KV blocks double-buffered in LDS via
// the Tensor Data Mover (TENSORcnt): block i+1 streams in behind block i's
// compute; TDM ops from one wave complete in order, so tensorcnt<=1 after
// issuing the next block guarantees the current block has landed.
// ---------------------------------------------------------------------------
__global__ void __launch_bounds__(128)
attn_kernel(const unsigned short* __restrict__ Qh,
            const unsigned short* __restrict__ KVh,
            float* __restrict__ Out)
{
    __shared__ __align__(16) unsigned short kvbuf[4][2][32*HDIM]; // 8KB / wave
    __shared__ __align__(16) unsigned short pbuf [4][16*32];      // 1KB / wave

    const int lane  = threadIdx.x & 31;
    const int wv    = threadIdx.x >> 5;
    const int wglob = blockIdx.x * 4 + wv;
    const int bh    = wglob >> 7;          // 0..31  (b*16 + h)
    const int qt    = wglob & 127;         // query tile
    const int half  = lane >> 4;
    const int l16   = lane & 15;

    const unsigned short* qbase = Qh  + ((size_t)bh * SLEN + qt*16) * HDIM;
    const unsigned short* kvb   = KVh + (size_t)bh * SLEN * HDIM;

    // Q A-fragments for dims [0,32) and [32,64)
    Frag q0, q1;
    {
        const unsigned short* qr = qbase + (size_t)l16 * HDIM;
        const int off = half ? 8 : 0;      // ushort offset
        q0.u4[0] = *(const ux4*)(qr + off);
        q0.u4[1] = *(const ux4*)(qr + off + 16);
        q1.u4[0] = *(const ux4*)(qr + 32 + off);
        q1.u4[1] = *(const ux4*)(qr + 32 + off + 16);
    }

    const fx8 zero8 = {0.f,0.f,0.f,0.f,0.f,0.f,0.f,0.f};
    fx8 oacc[4];
    #pragma unroll
    for (int t = 0; t < 4; ++t) oacc[t] = zero8;
    float m[8], l[8];
    #pragma unroll
    for (int r = 0; r < 8; ++r) { m[r] = -1e30f; l[r] = 0.f; }

    const float scale = 0.125f;            // 1/sqrt(64)
    unsigned short* pb = pbuf[wv];

#if USE_TDM
    const unsigned int lds_buf[2] = {
        (unsigned int)(uintptr_t)kvbuf[wv][0],
        (unsigned int)(uintptr_t)kvbuf[wv][1]
    };
    auto stage = [&](int jb, int buf) {
        tdm_stage_block(lds_buf[buf], kvb + (size_t)jb * HDIM);
    };
    auto wait_stage = [&](int outstanding) {
        if (outstanding)
            __builtin_amdgcn_s_wait_tensorcnt(1);   // imm: one TDM in flight
        else
            __builtin_amdgcn_s_wait_tensorcnt(0);   // imm: full drain
    };
#else
    const unsigned int lds_row[2] = {
        (unsigned int)(uintptr_t)(kvbuf[wv][0] + lane * HDIM),
        (unsigned int)(uintptr_t)(kvbuf[wv][1] + lane * HDIM)
    };
    auto stage = [&](int jb, int buf) {
        const unsigned short* g = kvb + (size_t)(jb + lane) * HDIM;
        #pragma unroll
        for (int t = 0; t < 8; ++t)
            async_copy_b128(lds_row[buf] + t*16, g + t*8);
    };
    auto wait_stage = [&](int outstanding) {
        if (outstanding)
            asm volatile("s_wait_asynccnt 0x8" ::: "memory");
        else
            asm volatile("s_wait_asynccnt 0x0" ::: "memory");
    };
#endif

    stage(0, 0);                           // prologue: block 0 -> buf 0

    for (int i = 0; i < SLEN/32; ++i) {
        const int jb = i * 32;
        unsigned short* kv = kvbuf[wv][i & 1];

        if (i + 1 < SLEN/32) {
            stage(jb + 32, (i + 1) & 1);   // stream next block behind compute
            wait_stage(1);                 // current block has landed
            if (jb + 64 < SLEN)            // warm L2 two blocks ahead
                __builtin_prefetch(kvb + (size_t)(jb + 64 + lane) * HDIM, 0, 1);
        } else {
            wait_stage(0);
        }

        // Scores: S[16q x 32k] as two 16x16 C-tiles, K read from LDS rows.
        fx8 s0 = zero8, s1 = zero8;
        #pragma unroll
        for (int n = 0; n < 2; ++n) {
            fx8& sacc = n ? s1 : s0;
            #pragma unroll
            for (int kk = 0; kk < 2; ++kk) {
                const unsigned short* kp =
                    kv + (n*16 + l16)*HDIM + kk*32 + (half ? 8 : 0);
                Frag kf;
                kf.u4[0] = *(const ux4*)(kp);
                kf.u4[1] = *(const ux4*)(kp + 16);
                sacc = __builtin_amdgcn_wmma_f32_16x16x32_bf16(
                    false, (kk ? q1.bf : q0.bf), false, kf.bf,
                    (short)0, sacc, false, false);
            }
        }

        // Online softmax across this 32-key block (row = 16 lanes of a half).
        #pragma unroll
        for (int r = 0; r < 8; ++r) {
            float a = s0[r] * scale;
            float b = s1[r] * scale;
            float v = fmaxf(a, b);
            v = fmaxf(v, __int_as_float(__builtin_amdgcn_ds_swizzle(__float_as_int(v), 0x041F)));
            v = fmaxf(v, __int_as_float(__builtin_amdgcn_ds_swizzle(__float_as_int(v), 0x081F)));
            v = fmaxf(v, __int_as_float(__builtin_amdgcn_ds_swizzle(__float_as_int(v), 0x101F)));
            v = fmaxf(v, __int_as_float(__builtin_amdgcn_ds_swizzle(__float_as_int(v), 0x201F)));
            const float mn    = fmaxf(m[r], v);
            const float alpha = __expf(m[r] - mn);
            m[r] = mn;
            const float p0 = __expf(a - mn);
            const float p1 = __expf(b - mn);
            float ps = p0 + p1;
            ps += __int_as_float(__builtin_amdgcn_ds_swizzle(__float_as_int(ps), 0x041F));
            ps += __int_as_float(__builtin_amdgcn_ds_swizzle(__float_as_int(ps), 0x081F));
            ps += __int_as_float(__builtin_amdgcn_ds_swizzle(__float_as_int(ps), 0x101F));
            ps += __int_as_float(__builtin_amdgcn_ds_swizzle(__float_as_int(ps), 0x201F));
            l[r] = l[r] * alpha + ps;
            #pragma unroll
            for (int t = 0; t < 4; ++t) oacc[t][r] *= alpha;
            const int mrow = r + half*8;
            pb[mrow*32 + l16]      = f2bf(p0);
            pb[mrow*32 + l16 + 16] = f2bf(p1);
        }

        // P (16x32) as A-fragment via LDS round-trip.
        Frag pf;
        {
            const unsigned short* pp = pb + l16*32 + (half ? 16 : 0);
            pf.u4[0] = *(const ux4*)(pp);
            pf.u4[1] = *(const ux4*)(pp + 8);
        }

        // O += P @ V  (V == KV block; B-fragments gathered column-wise)
        #pragma unroll
        for (int t = 0; t < 4; ++t) {
            Frag vf;
            const int dim = t*16 + l16;
            const int kh  = half ? 16 : 0;
            #pragma unroll
            for (int p = 0; p < 8; ++p) {
                unsigned int lo = kv[(kh + 2*p    )*HDIM + dim];
                unsigned int hi = kv[(kh + 2*p + 1)*HDIM + dim];
                vf.u[p] = lo | (hi << 16);
            }
            oacc[t] = __builtin_amdgcn_wmma_f32_16x16x32_bf16(
                false, pf.bf, false, vf.bf, (short)0, oacc[t], false, false);
        }
    }

    // Epilogue: out[b, s, h*64 + i] = O / l   (f32, 'B n S d -> B S (n d)')
    const int bidx = bh >> 4;
    const int h    = bh & 15;
    #pragma unroll
    for (int t = 0; t < 4; ++t) {
        #pragma unroll
        for (int r = 0; r < 8; ++r) {
            const int srow = qt*16 + r + half*8;
            const int ch   = h*HDIM + t*16 + l16;
            Out[((size_t)(bidx*SLEN + srow)) * DMODEL + ch] = oacc[t][r] / l[r];
        }
    }
}

// ---------------------------------------------------------------------------
extern "C" void kernel_launch(void* const* d_in, const int* in_sizes, int n_in,
                              void* d_out, int out_size, void* d_ws, size_t ws_size,
                              hipStream_t stream) {
    const float* x  = (const float*)d_in[0];
    const float* Wq = (const float*)d_in[1];
    const float* bq = (const float*)d_in[2];
    const float* Wv = (const float*)d_in[3];
    const float* bv = (const float*)d_in[4];
    float* out = (float*)d_out;

    // Workspace layout (bf16/ushort units):
    //   xb  [4096][1024]   8.4 MB
    //   Wqb [1024][1024]   2.1 MB
    //   Wvb [1024][1024]   2.1 MB
    //   Qh  [32][2048][64] 8.4 MB   (head-major)
    //   KVh [32][2048][64] 8.4 MB   (head-major; serves both K and V)
    unsigned short* xb  = (unsigned short*)d_ws;
    unsigned short* Wqb = xb  + (size_t)BATCH * SLEN * DMODEL;
    unsigned short* Wvb = Wqb + (size_t)DMODEL * DMODEL;
    unsigned short* Qh  = Wvb + (size_t)DMODEL * DMODEL;
    unsigned short* KVh = Qh  + (size_t)BATCH * NHEADS * SLEN * HDIM;

    // f32 -> bf16 conversion passes (packed, one touch per element).
    cvt_bf16_kernel<<<2048, 256, 0, stream>>>(x,  xb,  BATCH*SLEN*DMODEL/8);
    cvt_bf16_kernel<<<512,  256, 0, stream>>>(Wq, Wqb, DMODEL*DMODEL/8);
    cvt_bf16_kernel<<<512,  256, 0, stream>>>(Wv, Wvb, DMODEL*DMODEL/8);

    // 4096 row-tiles*colgroups / 8 waves per 256-thread block = 512 blocks.
    proj_kernel<<<512, 256, 0, stream>>>(xb, Wqb, bq, Qh);
    proj_kernel<<<512, 256, 0, stream>>>(xb, Wvb, bv, KVh);
    // 32 (b,h) * 128 query tiles = 4096 waves / 4 per 128-thread block.
    attn_kernel<<<1024, 128, 0, stream>>>(Qh, KVh, out);
}
